// GraphAttentionLayer_12549894439564
// MI455X (gfx1250) — matrix operand order repaired and checked by reading, compiled, not run on hardware
//
#include <hip/hip_runtime.h>
#include <hip/hip_bf16.h>
#include <math.h>

#define BB   16
#define NN   2048
#define DD   128
#define OPE  64

typedef __attribute__((ext_vector_type(16))) __bf16       v16bf;
typedef __attribute__((ext_vector_type(8)))  float        v8f;
typedef __attribute__((ext_vector_type(4)))  unsigned int u32x4;
typedef __attribute__((ext_vector_type(4)))  float        f32x4;

union BF16Frag {
    v16bf v;
    u32x4 q[2];
    unsigned short h[16];
};

__device__ __forceinline__ unsigned short f2bf(float f) {
    union { float f; unsigned u; } a; a.f = f;
    unsigned r = a.u + 0x7FFFu + ((a.u >> 16) & 1u);   // round-to-nearest-even
    return (unsigned short)(r >> 16);
}

// CDNA5 async memory->LDS copy (16B per lane), tracked by ASYNCcnt.
__device__ __forceinline__ void async_ld16(unsigned int lds_off, const void* gptr) {
    asm volatile("global_load_async_to_lds_b128 %0, %1, off"
                 :: "v"(lds_off), "v"((unsigned long long)(size_t)gptr)
                 : "memory");
}
__device__ __forceinline__ void wait_async_le4() {
    asm volatile("s_wait_asynccnt 0x4" ::: "memory");
}
__device__ __forceinline__ void wait_async_0() {
    asm volatile("s_wait_asynccnt 0x0" ::: "memory");
}

// ---------------------------------------------------------------------------
// Kernel 0: transpose weights to out-major bf16; fold a_w into Wk columns.
// Wt layout: [m][out][in] bf16, m: 0=Q, 1=K' (a_w folded), 2=V.
// ---------------------------------------------------------------------------
__global__ void prep_w(const float* __restrict__ Wq, const float* __restrict__ Wk,
                       const float* __restrict__ Wv, const float* __restrict__ a_w,
                       unsigned short* __restrict__ Wt) {
    int idx = blockIdx.x * blockDim.x + threadIdx.x;
    if (idx >= 3 * DD * DD) return;
    int m = idx >> 14;          // 128*128 = 16384
    int r = idx & 16383;
    int o = r >> 7;
    int i = r & 127;
    float w;
    if (m == 0)      w = Wq[i * DD + o];
    else if (m == 1) w = Wk[i * DD + o] * a_w[o];
    else             w = Wv[i * DD + o];
    Wt[idx] = f2bf(w);
}

// ---------------------------------------------------------------------------
// Kernel 1: Q/K'/V projections with bf16 WMMA.
// Q: row-major [b][e][d] bf16, K': row-major [b][l][d] bf16,
// V: transposed [b][d][l] bf16.
// ---------------------------------------------------------------------------
__global__ __launch_bounds__(256) void proj_qkv(const float* __restrict__ h,
                                                const unsigned short* __restrict__ Wt,
                                                unsigned short* __restrict__ Qw,
                                                unsigned short* __restrict__ Kw,
                                                unsigned short* __restrict__ Vw) {
    int lane = threadIdx.x & 31;
    int wave = threadIdx.x >> 5;
    int tile = blockIdx.x * 8 + wave;        // 0 .. B*N/16-1
    int b    = tile >> 7;                    // 128 tiles per batch
    int row0 = (tile & 127) << 4;
    int hi = lane >> 4, ln = lane & 15;

    // A-fragments of h (16 rows x 128 K), f32 -> bf16.
    const float* hrow = h + ((size_t)(b * NN + row0 + ln)) * DD;
    BF16Frag A[4];
#pragma unroll
    for (int c = 0; c < 4; ++c) {
        int off = 32 * c + hi * 8;
        f32x4 f0 = *(const f32x4*)(hrow + off);
        f32x4 f1 = *(const f32x4*)(hrow + off + 4);
        f32x4 f2 = *(const f32x4*)(hrow + off + 16);
        f32x4 f3 = *(const f32x4*)(hrow + off + 20);
#pragma unroll
        for (int k = 0; k < 4; ++k) {
            A[c].h[k]      = f2bf(f0[k]);
            A[c].h[4 + k]  = f2bf(f1[k]);
            A[c].h[8 + k]  = f2bf(f2[k]);
            A[c].h[12 + k] = f2bf(f3[k]);
        }
    }

    v8f zero = {0.f, 0.f, 0.f, 0.f, 0.f, 0.f, 0.f, 0.f};
#pragma unroll
    for (int m = 0; m < 3; ++m) {
        const unsigned short* Wm = Wt + m * (DD * DD);
#pragma unroll
        for (int j = 0; j < 8; ++j) {
            // preload all 4 B-fragments, then chain the WMMAs
            BF16Frag Bf[4];
#pragma unroll
            for (int c = 0; c < 4; ++c) {
                const unsigned short* bp = Wm + (16 * j + ln) * DD + 32 * c + hi * 8;
                Bf[c].q[0] = *(const u32x4*)bp;
                Bf[c].q[1] = *(const u32x4*)(bp + 16);
            }
            v8f C = zero;
#pragma unroll
            for (int c = 0; c < 4; ++c)
                C = __builtin_amdgcn_wmma_f32_16x16x32_bf16(false, A[c].v, false, Bf[c].v,
                                                            (short)0, C, false, false);
            // C layout: lane = column (d = 16j+ln), VGPR r = row e = r + 8*hi.
            if (m == 2) {
#pragma unroll
                for (int r = 0; r < 8; ++r) {
                    int e = row0 + r + 8 * hi;
                    int d = 16 * j + ln;
                    Vw[((size_t)(b * DD + d)) * NN + e] = f2bf(C[r]);   // transposed
                }
            } else {
                unsigned short* dst = (m == 0) ? Qw : Kw;
#pragma unroll
                for (int r = 0; r < 8; ++r) {
                    int e = row0 + r + 8 * hi;
                    int d = 16 * j + ln;
                    dst[((size_t)(b * NN + e)) * DD + d] = f2bf(C[r]);
                }
            }
        }
    }
}

// ---------------------------------------------------------------------------
// Kernel 2: gate = sigmoid(op_emb @ op_W + op_b). One block per (b,n) row.
// ---------------------------------------------------------------------------
__global__ __launch_bounds__(128) void gate_kernel(const float* __restrict__ op_emb,
                                                   const float* __restrict__ op_W,
                                                   const float* __restrict__ op_b,
                                                   float* __restrict__ gate) {
    __shared__ float se[OPE];
    int bn = blockIdx.x;
    int d  = threadIdx.x;
    if (d < OPE) se[d] = op_emb[(size_t)bn * OPE + d];
    __syncthreads();
    float acc = op_b[d];
#pragma unroll 8
    for (int k = 0; k < OPE; ++k) acc += se[k] * op_W[k * DD + d];
    gate[(size_t)bn * DD + d] = 1.f / (1.f + __expf(-acc));
}

// ---------------------------------------------------------------------------
// Kernel 3: fused flash-style attention + gate + LayerNorm.
// Double-buffered K'/V tiles staged with GLOBAL_LOAD_ASYNC_TO_LDS_B128
// (ASYNCcnt), adj streamed from HBM exactly once with prefetch.
// ---------------------------------------------------------------------------
__global__ __launch_bounds__(256) void attn_kernel(const unsigned short* __restrict__ Qw,
                                                   const unsigned short* __restrict__ Kw,
                                                   const unsigned short* __restrict__ Vw,
                                                   const float* __restrict__ adj,
                                                   const float* __restrict__ gate,
                                                   const float* __restrict__ ln_g,
                                                   const float* __restrict__ ln_b,
                                                   float* __restrict__ out) {
    __shared__ unsigned short Klds[2][32 * DD];    // K' tile, row-major [l][d]
    __shared__ unsigned short Vlds[2][DD * 32];    // V tile, transposed [d][l]
    __shared__ unsigned short Plds[8 * 16 * 32];   // per-wave P scratch [e][l]

    int tid  = threadIdx.x;
    int lane = tid & 31;
    int wave = tid >> 5;
    int b    = blockIdx.y;
    int row0 = blockIdx.x * 128 + wave * 16;
    int hi = lane >> 4, ln = lane & 15;
    const float inv_sqrt_d = 0.08838834764831845f;  // 1/sqrt(128)

    // Async staging of one 32-l tile: 4 x 16B per thread (2 K chunks + 2 V chunks).
    auto stage = [&](int buf, int l0) {
#pragma unroll
        for (int s = 0; s < 2; ++s) {
            int id = tid + s * 256;                 // 0..511
            int l = id >> 4, p = id & 15;           // K: 16 chunks per 256B row
            const unsigned short* gk = Kw + ((size_t)(b * NN + l0 + l)) * DD + p * 8;
            async_ld16((unsigned int)(size_t)&Klds[buf][l * DD + p * 8], gk);
            int d = id >> 2, q = id & 3;            // V: 4 chunks per 64B row
            const unsigned short* gv = Vw + ((size_t)(b * DD + d)) * NN + l0 + q * 8;
            async_ld16((unsigned int)(size_t)&Vlds[buf][d * 32 + q * 8], gv);
        }
    };

    // Resident Q A-fragments (16 e-rows x 128 d).
    const unsigned short* qrow = Qw + ((size_t)(b * NN + row0 + ln)) * DD;
    BF16Frag QA[4];
#pragma unroll
    for (int c = 0; c < 4; ++c) {
        int off = 32 * c + hi * 8;
        QA[c].q[0] = *(const u32x4*)(qrow + off);
        QA[c].q[1] = *(const u32x4*)(qrow + off + 16);
    }

    v8f zero = {0.f, 0.f, 0.f, 0.f, 0.f, 0.f, 0.f, 0.f};
    v8f agg[8];
#pragma unroll
    for (int j = 0; j < 8; ++j) agg[j] = zero;
    float mrow[8], srow[8];
#pragma unroll
    for (int r = 0; r < 8; ++r) { mrow[r] = -1e30f; srow[r] = 0.f; }

    unsigned short* Pw = Plds + wave * (16 * 32);

    stage(0, 0);                                    // prologue: tile 0 in flight
    for (int it = 0; it < NN / 32; ++it) {
        int l0  = it * 32;
        int cur = it & 1;
        if (it + 1 < NN / 32) {                     // next tile into alternate buffer
            stage(cur ^ 1, l0 + 32);
            wait_async_le4();                       // retire current tile's 4 ops
        } else {
            wait_async_0();
        }
        __syncthreads();                            // all waves' tile data landed

        // scores for two 16-wide l sub-tiles
        float x[2][8];
#pragma unroll
        for (int t = 0; t < 2; ++t) {
            BF16Frag Bf[4];                         // B = K'^T: lane col = l, K rows = d
#pragma unroll
            for (int c = 0; c < 4; ++c) {
                const unsigned short* bp = &Klds[cur][(16 * t + ln) * DD + 32 * c + hi * 8];
                Bf[c].q[0] = *(const u32x4*)bp;
                Bf[c].q[1] = *(const u32x4*)(bp + 16);
            }
            v8f S = zero;
#pragma unroll
            for (int c = 0; c < 4; ++c)
                S = __builtin_amdgcn_wmma_f32_16x16x32_bf16(false, QA[c].v, false, Bf[c].v,
                                                            (short)0, S, false, false);
            if (l0 + 32 < NN) {                     // prefetch next adj chunk (HBM stream)
                int e0 = row0 + 8 * hi;
                __builtin_prefetch(&adj[((size_t)(b * NN + e0)) * NN + l0 + 32 + 16 * t + ln], 0, 1);
            }
#pragma unroll
            for (int r = 0; r < 8; ++r) {
                int e = row0 + r + 8 * hi;
                float sc = S[r] * inv_sqrt_d;
                sc = sc > 0.f ? sc : 0.2f * sc;                 // leaky relu
                float a = adj[((size_t)(b * NN + e)) * NN + l0 + 16 * t + ln];
                x[t][r] = sc * a;                               // multiplicative mask
            }
        }

        // online softmax update (row = (r, half); reduce across 16-lane half)
        float cfac[8];
#pragma unroll
        for (int r = 0; r < 8; ++r) {
            float tmax = fmaxf(x[0][r], x[1][r]);
#pragma unroll
            for (int m = 1; m < 16; m <<= 1) tmax = fmaxf(tmax, __shfl_xor(tmax, m, 32));
            float mnew = fmaxf(mrow[r], tmax);
            cfac[r] = __expf(mrow[r] - mnew);
            mrow[r] = mnew;
            float p0 = __expf(x[0][r] - mnew);
            float p1 = __expf(x[1][r] - mnew);
            x[0][r] = p0; x[1][r] = p1;
            float rs = p0 + p1;
#pragma unroll
            for (int m = 1; m < 16; m <<= 1) rs += __shfl_xor(rs, m, 32);
            srow[r] = srow[r] * cfac[r] + rs;
        }
#pragma unroll
        for (int j = 0; j < 8; ++j)
#pragma unroll
            for (int r = 0; r < 8; ++r)
                agg[j][r] = agg[j][r] * cfac[r];

        // re-layout P (C layout -> A layout) through per-wave LDS scratch
#pragma unroll
        for (int t = 0; t < 2; ++t)
#pragma unroll
            for (int r = 0; r < 8; ++r)
                Pw[(r + 8 * hi) * 32 + 16 * t + ln] = f2bf(x[t][r]);
        BF16Frag PA;
        {
            const unsigned short* pr = Pw + ln * 32 + hi * 8;
            PA.q[0] = *(const u32x4*)pr;
            PA.q[1] = *(const u32x4*)(pr + 16);
        }

        // agg += P @ V   (8 d-tiles)
#pragma unroll
        for (int j = 0; j < 8; ++j) {
            BF16Frag Vf;                            // B = V: lane col = d, K rows = l
            const unsigned short* vp = &Vlds[cur][(16 * j + ln) * 32 + hi * 8];
            Vf.q[0] = *(const u32x4*)vp;
            Vf.q[1] = *(const u32x4*)(vp + 16);
            agg[j] = __builtin_amdgcn_wmma_f32_16x16x32_bf16(false, PA.v, false, Vf.v,
                                                             (short)0, agg[j], false, false);
        }
        __syncthreads();                            // protect buffer reuse next iter
    }

    // epilogue: 1/rowsum, gate, LayerNorm, store
    float invs[8];
#pragma unroll
    for (int r = 0; r < 8; ++r) invs[r] = 1.f / srow[r];
    float g[8], bia[8];
#pragma unroll
    for (int j = 0; j < 8; ++j) { g[j] = ln_g[16 * j + ln]; bia[j] = ln_b[16 * j + ln]; }

    float psum[8], psq[8];
#pragma unroll
    for (int r = 0; r < 8; ++r) { psum[r] = 0.f; psq[r] = 0.f; }
#pragma unroll
    for (int j = 0; j < 8; ++j) {
#pragma unroll
        for (int r = 0; r < 8; ++r) {
            int e = row0 + r + 8 * hi;
            int d = 16 * j + ln;
            float v = agg[j][r] * invs[r];
            v *= gate[((size_t)(b * NN + e)) * DD + d];
            agg[j][r] = v;
            psum[r] += v;
            psq[r]  += v * v;
        }
    }
    float mean[8], rstd[8];
#pragma unroll
    for (int r = 0; r < 8; ++r) {
        float s1 = psum[r], s2 = psq[r];
#pragma unroll
        for (int m = 1; m < 16; m <<= 1) {
            s1 += __shfl_xor(s1, m, 32);
            s2 += __shfl_xor(s2, m, 32);
        }
        mean[r] = s1 * (1.f / 128.f);
        float var = s2 * (1.f / 128.f) - mean[r] * mean[r];
        rstd[r] = rsqrtf(var + 1e-5f);
    }
#pragma unroll
    for (int j = 0; j < 8; ++j) {
#pragma unroll
        for (int r = 0; r < 8; ++r) {
            int e = row0 + r + 8 * hi;
            int d = 16 * j + ln;
            out[((size_t)(b * NN + e)) * DD + d] =
                (agg[j][r] - mean[r]) * rstd[r] * g[j] + bia[j];
        }
    }
}

// ---------------------------------------------------------------------------
extern "C" void kernel_launch(void* const* d_in, const int* in_sizes, int n_in,
                              void* d_out, int out_size, void* d_ws, size_t ws_size,
                              hipStream_t stream) {
    (void)in_sizes; (void)n_in; (void)out_size; (void)ws_size;
    const float* h      = (const float*)d_in[0];
    const float* adj    = (const float*)d_in[1];
    const float* op_emb = (const float*)d_in[2];
    const float* Wk     = (const float*)d_in[3];
    const float* Wq     = (const float*)d_in[4];
    const float* Wv     = (const float*)d_in[5];
    const float* a_w    = (const float*)d_in[6];
    const float* op_W   = (const float*)d_in[7];
    const float* op_b   = (const float*)d_in[8];
    const float* ln_g   = (const float*)d_in[9];
    const float* ln_b   = (const float*)d_in[10];
    float* out = (float*)d_out;

    char* ws = (char*)d_ws;
    const size_t WT_BYTES  = 3 * DD * DD * sizeof(unsigned short);          // 96 KiB
    const size_t QKV_BYTES = (size_t)BB * NN * DD * sizeof(unsigned short); // 8 MiB each
    unsigned short* Wt = (unsigned short*)(ws);
    unsigned short* Qw = (unsigned short*)(ws + WT_BYTES);
    unsigned short* Kw = (unsigned short*)(ws + WT_BYTES + QKV_BYTES);
    unsigned short* Vw = (unsigned short*)(ws + WT_BYTES + 2 * QKV_BYTES);
    float* gate        = (float*)(ws + WT_BYTES + 3 * QKV_BYTES);           // 16 MiB

    hipLaunchKernelGGL(prep_w, dim3((3 * DD * DD + 255) / 256), dim3(256), 0, stream,
                       Wq, Wk, Wv, a_w, Wt);
    hipLaunchKernelGGL(proj_qkv, dim3(BB * NN / 16 / 8), dim3(256), 0, stream,
                       h, Wt, Qw, Kw, Vw);
    hipLaunchKernelGGL(gate_kernel, dim3(BB * NN), dim3(128), 0, stream,
                       op_emb, op_W, op_b, gate);
    hipLaunchKernelGGL(attn_kernel, dim3(NN / 128, BB), dim3(256), 0, stream,
                       Qw, Kw, Vw, adj, gate, ln_g, ln_b, out);
}
